// PhysicalStreamEncoder_27582279975544
// MI455X (gfx1250) — matrix operand order repaired and checked by reading, not compiled, not run on hardware
//
#include <hip/hip_runtime.h>
#include <hip/hip_bf16.h>

typedef __attribute__((ext_vector_type(16))) __bf16 v16bf;
typedef __attribute__((ext_vector_type(8)))  float  v8f;

#define DD    512
#define HH    8
#define HDIM  64
#define NNTOK 1024
#define BBAT  8
#define LLAY  4
#define DFFN  2048
#define NBINS 16
#define MROWS (BBAT * NNTOK)   // 8192

__device__ __forceinline__ unsigned short f2bf(float f) {
    unsigned int u = __float_as_uint(f);
    unsigned int r = (u + 0x7FFFu + ((u >> 16) & 1u)) >> 16;
    return (unsigned short)r;
}
__device__ __forceinline__ float bf2f(unsigned short s) {
    return __uint_as_float(((unsigned int)s) << 16);
}
__device__ __forceinline__ __bf16 bfbits(unsigned short s) {
    union { unsigned short u; __bf16 b; } v; v.u = s; return v.b;
}

// ---------------------------------------------------------------------------
// fp32 -> bf16 conversion (plain)
// ---------------------------------------------------------------------------
__global__ void k_cvt(const float* __restrict__ src, unsigned short* __restrict__ dst, int n) {
    int i = blockIdx.x * 256 + threadIdx.x;
    if (i < n) dst[i] = f2bf(src[i]);
}

// ---------------------------------------------------------------------------
// fp32 [L][K][Ncol] -> bf16 transposed [L][Ncol][K], tiled through LDS
// grid: (K/32, Ncol/32, L), block 256
// ---------------------------------------------------------------------------
__global__ __launch_bounds__(256)
void k_cvtT(const float* __restrict__ src, unsigned short* __restrict__ dst,
            int K, int Ncol) {
    __shared__ unsigned short tile[32][33];
    const int t = threadIdx.x;
    const int kr0 = blockIdx.x * 32;
    const int c0  = blockIdx.y * 32;
    const size_t l = blockIdx.z;
#pragma unroll
    for (int e = 0; e < 4; ++e) {
        const int idx = t + e * 256;
        const int r = idx >> 5, c = idx & 31;
        tile[c][r] = f2bf(src[(l * K + kr0 + r) * (size_t)Ncol + c0 + c]);
    }
    __syncthreads();
#pragma unroll
    for (int e = 0; e < 4; ++e) {
        const int idx = t + e * 256;
        const int r = idx >> 5, c = idx & 31;   // r: col-dim, c: k-dim
        dst[(l * Ncol + c0 + r) * (size_t)K + kr0 + c] = tile[r][c];
    }
}

// ---------------------------------------------------------------------------
// input projection (4 -> 512) + ReLU + LayerNorm, writes fp32 residual stream
// ---------------------------------------------------------------------------
__global__ __launch_bounds__(256)
void k_inproj(const float* __restrict__ x, const float* __restrict__ w,
              const float* __restrict__ bi, const float* __restrict__ g,
              const float* __restrict__ be, float* __restrict__ h) {
    __shared__ float r1[256], r2[256];
    const int row = blockIdx.x;
    const int t = threadIdx.x;
    const float f0 = x[row * 4 + 0], f1 = x[row * 4 + 1];
    const float f2 = x[row * 4 + 2], f3 = x[row * 4 + 3];
    float o[2];
#pragma unroll
    for (int i = 0; i < 2; ++i) {
        const int c = t + i * 256;
        float v = f0 * w[c] + f1 * w[DD + c] + f2 * w[2 * DD + c] + f3 * w[3 * DD + c] + bi[c];
        o[i] = fmaxf(v, 0.0f);
    }
    r1[t] = o[0] + o[1];
    r2[t] = o[0] * o[0] + o[1] * o[1];
    __syncthreads();
    for (int s = 128; s > 0; s >>= 1) {
        if (t < s) { r1[t] += r1[t + s]; r2[t] += r2[t + s]; }
        __syncthreads();
    }
    const float mean = r1[0] * (1.0f / DD);
    const float var  = r2[0] * (1.0f / DD) - mean * mean;
    const float rstd = rsqrtf(var + 1e-5f);
#pragma unroll
    for (int i = 0; i < 2; ++i) {
        const int c = t + i * 256;
        h[(size_t)row * DD + c] = (o[i] - mean) * rstd * g[c] + be[c];
    }
}

// ---------------------------------------------------------------------------
// LayerNorm (fp32 in) -> bf16 out
// ---------------------------------------------------------------------------
__global__ __launch_bounds__(256)
void k_ln(const float* __restrict__ x, const float* __restrict__ g,
          const float* __restrict__ b, unsigned short* __restrict__ out) {
    __shared__ float r1[256], r2[256];
    const int row = blockIdx.x;
    const int t = threadIdx.x;
    const float v0 = x[(size_t)row * DD + t];
    const float v1 = x[(size_t)row * DD + t + 256];
    r1[t] = v0 + v1;
    r2[t] = v0 * v0 + v1 * v1;
    __syncthreads();
    for (int s = 128; s > 0; s >>= 1) {
        if (t < s) { r1[t] += r1[t + s]; r2[t] += r2[t + s]; }
        __syncthreads();
    }
    const float mean = r1[0] * (1.0f / DD);
    const float var  = r2[0] * (1.0f / DD) - mean * mean;
    const float rstd = rsqrtf(var + 1e-5f);
    out[(size_t)row * DD + t]       = f2bf((v0 - mean) * rstd * g[t] + b[t]);
    out[(size_t)row * DD + t + 256] = f2bf((v1 - mean) * rstd * g[t + 256] + b[t + 256]);
}

// ---------------------------------------------------------------------------
// per-batch max of distance matrix (distances >= 0 -> uint-bit atomicMax)
// ---------------------------------------------------------------------------
__global__ void k_maxinit(unsigned int* bmax) {
    if (threadIdx.x < BBAT) bmax[threadIdx.x] = 0u;
}

__global__ __launch_bounds__(256)
void k_max(const float* __restrict__ dm, unsigned int* __restrict__ bmax) {
    __shared__ float r[256];
    const int b = blockIdx.y;
    const size_t base = (size_t)b * NNTOK * NNTOK + (size_t)blockIdx.x * 16384;
    float m = 0.0f;
    for (int i = threadIdx.x; i < 16384; i += 256) m = fmaxf(m, dm[base + i]);
    r[threadIdx.x] = m;
    __syncthreads();
    for (int s = 128; s > 0; s >>= 1) {
        if (threadIdx.x < s) r[threadIdx.x] = fmaxf(r[threadIdx.x], r[threadIdx.x + s]);
        __syncthreads();
    }
    if (threadIdx.x == 0) atomicMax(&bmax[b], __float_as_uint(r[0]));
}

// ---------------------------------------------------------------------------
// binned distances, transposed to [b][key][row], pad-mask folded in (bin 16)
// grid: (N/64 row-tiles, N/64 key-tiles, B), block 256. LDS tiled transpose.
// ---------------------------------------------------------------------------
__global__ __launch_bounds__(256)
void k_binsT(const float* __restrict__ dm, const unsigned int* __restrict__ bmax,
             const unsigned char* __restrict__ mask, unsigned char* __restrict__ binsT) {
    __shared__ unsigned char tile[64][65];
    const int t = threadIdx.x;
    const int r0 = blockIdx.x * 64;
    const int k0 = blockIdx.y * 64;
    const int b  = blockIdx.z;
    const float md = fmaxf(__uint_as_float(bmax[b]), 1e-6f);
    const float sc = (float)(NBINS - 1) / md;
#pragma unroll
    for (int e = 0; e < 16; ++e) {
        const int idx = t + e * 256;
        const int r = idx >> 6, c = idx & 63;
        const int row = r0 + r, key = k0 + c;
        int bi = (int)(dm[((size_t)b << 20) + ((size_t)row << 10) + key] * sc);  // trunc
        bi = min(max(bi, 0), NBINS - 1);
        const bool ok = (mask[b * NNTOK + row] != 0) && (mask[b * NNTOK + key] != 0);
        tile[c][r] = (unsigned char)(ok ? bi : NBINS);
    }
    __syncthreads();
#pragma unroll
    for (int e = 0; e < 16; ++e) {
        const int idx = t + e * 256;
        const int kk = idx >> 6, rr = idx & 63;
        binsT[((size_t)b * NNTOK + k0 + kk) * NNTOK + r0 + rr] = tile[kk][rr];
    }
}

// ---------------------------------------------------------------------------
// tiled bf16 WMMA GEMM: C(MxNcol) = A(MxK) @ W(KxNcol) + bias, W given
// TRANSPOSED bf16 [Ncol][K]. Block 256 = 8 waves, BM=BN=128, BK=32,
// wave computes 64x32. Ping-pong double-buffered LDS: next tile's global
// loads are issued before the WMMAs of the current tile, one barrier/step.
// ---------------------------------------------------------------------------
#define BM 128
#define BN 128
#define BK 32

enum { EPI_QKV = 0, EPI_GELU = 1, EPI_RES = 2, EPI_OUT = 3 };

template <int EPI>
__global__ __launch_bounds__(256)
void k_gemm(const unsigned short* __restrict__ A,
            const unsigned short* __restrict__ BwT,  // [Ncol][K]
            const float* __restrict__ bias,
            int Ncol, int K,
            float* __restrict__ resF,
            unsigned short* __restrict__ outB,
            unsigned short* __restrict__ qb,
            unsigned short* __restrict__ kb,
            unsigned short* __restrict__ vtb) {
    __shared__ unsigned short As[2][BM * BK];   // [m][k]
    __shared__ unsigned short Bs[2][BN * BK];   // [col][k]  (fragment-major)
    const int tid  = threadIdx.x;
    const int wave = tid >> 5;
    const int lane = tid & 31;
    const int half = lane >> 4;
    const int lidx = lane & 15;
    const int tn = blockIdx.x * BN;
    const int tm = blockIdx.y * BM;
    const int wmo = (wave >> 2) * 64;   // 2 waves in M
    const int wno = (wave & 3) * 32;    // 4 waves in N

    v8f acc[4][2];
#pragma unroll
    for (int mi = 0; mi < 4; ++mi)
#pragma unroll
        for (int ni = 0; ni < 2; ++ni)
#pragma unroll
            for (int r = 0; r < 8; ++r) acc[mi][ni][r] = 0.0f;

    const int lr = (tid * 16) >> 5;   // tile row (A: m, B: col) this thread loads
    const int lc = (tid * 16) & 31;   // k offset (0 or 16)
    const unsigned short* arow = A   + (size_t)(tm + lr) * K + lc;
    const unsigned short* brow = BwT + (size_t)(tn + lr) * K + lc;

    // prime buffer 0
    {
        uint4 a0 = *(const uint4*)(arow);
        uint4 a1 = *(const uint4*)(arow + 8);
        uint4 b0 = *(const uint4*)(brow);
        uint4 b1 = *(const uint4*)(brow + 8);
        *(uint4*)&As[0][lr * BK + lc]     = a0;
        *(uint4*)&As[0][lr * BK + lc + 8] = a1;
        *(uint4*)&Bs[0][lr * BK + lc]     = b0;
        *(uint4*)&Bs[0][lr * BK + lc + 8] = b1;
    }
    __syncthreads();

    int buf = 0;
    for (int k0 = 0; k0 < K; k0 += BK, buf ^= 1) {
        // 1) issue next tile's global loads early (latency overlaps WMMAs)
        uint4 a0, a1, b0, b1;
        const bool more = (k0 + BK) < K;
        if (more) {
            a0 = *(const uint4*)(arow + k0 + BK);
            a1 = *(const uint4*)(arow + k0 + BK + 8);
            b0 = *(const uint4*)(brow + k0 + BK);
            b1 = *(const uint4*)(brow + k0 + BK + 8);
        }

        // 2) fragments from current buffer + WMMAs
        v16bf afr[4];
        v16bf bfr[2];
#pragma unroll
        for (int mi = 0; mi < 4; ++mi) {
            // A fragment: lanes 0-15 K=0..15, lanes 16-31 K=16..31
            const unsigned short* ap = &As[buf][(wmo + mi * 16 + lidx) * BK + 16 * half];
#pragma unroll
            for (int j = 0; j < 8; ++j) afr[mi][j] = bfbits(ap[j]);
#pragma unroll
            for (int j = 0; j < 8; ++j) afr[mi][8 + j] = bfbits(ap[8 + j]);
        }
#pragma unroll
        for (int ni = 0; ni < 2; ++ni) {
            // B fragment: lane holds col, lanes 0-15 K=0..15, lanes 16-31 K=16..31
            const unsigned short* bp = &Bs[buf][(wno + ni * 16 + lidx) * BK + 16 * half];
#pragma unroll
            for (int j = 0; j < 16; ++j) bfr[ni][j] = bfbits(bp[j]);
        }
#pragma unroll
        for (int mi = 0; mi < 4; ++mi)
#pragma unroll
            for (int ni = 0; ni < 2; ++ni)
                acc[mi][ni] = __builtin_amdgcn_wmma_f32_16x16x32_bf16(
                    false, afr[mi], false, bfr[ni], (short)0, acc[mi][ni], false, false);

        // 3) stage next tile into the other buffer
        if (more) {
            const int nb = buf ^ 1;
            *(uint4*)&As[nb][lr * BK + lc]     = a0;
            *(uint4*)&As[nb][lr * BK + lc + 8] = a1;
            *(uint4*)&Bs[nb][lr * BK + lc]     = b0;
            *(uint4*)&Bs[nb][lr * BK + lc + 8] = b1;
        }
        // 4) single barrier per step
        __syncthreads();
    }

#pragma unroll
    for (int mi = 0; mi < 4; ++mi) {
#pragma unroll
        for (int ni = 0; ni < 2; ++ni) {
#pragma unroll
            for (int r = 0; r < 8; ++r) {
                const int grow = tm + wmo + mi * 16 + r + 8 * half;  // C: M = r + 8*half
                const int gcol = tn + wno + ni * 16 + lidx;          // N = lane&15
                float v = acc[mi][ni][r] + bias[gcol];
                if (EPI == EPI_QKV) {
                    const int j = gcol >> 9;        // 0:q 1:k 2:v
                    const int dd = gcol & 511;
                    const int head = dd >> 6;
                    const int hd = dd & 63;
                    const int bidx = grow >> 10;
                    const int n = grow & 1023;
                    if (j == 2) {  // V stored transposed: [b,h,hd,key]
                        vtb[(((size_t)(bidx * HH + head)) * HDIM + hd) * NNTOK + n] = f2bf(v);
                    } else {
                        unsigned short* dst = (j == 0) ? qb : kb;
                        dst[(((size_t)(bidx * HH + head)) * NNTOK + n) * HDIM + hd] = f2bf(v);
                    }
                } else if (EPI == EPI_GELU) {
                    v = 0.5f * v * (1.0f + erff(v * 0.70710678118654752f));
                    outB[(size_t)grow * Ncol + gcol] = f2bf(v);
                } else if (EPI == EPI_RES) {
                    resF[(size_t)grow * DD + gcol] += v;   // Ncol == 512 here
                } else {
                    resF[(size_t)grow * Ncol + gcol] = v;
                }
            }
        }
    }
}

// ---------------------------------------------------------------------------
// attention: one wave handles 16 query rows of one (b,h). Branch-free inner
// loop: bins (transposed, mask-folded) give a single b64 load per tile; bias
// via 32-entry LDS table (entries 16..31 = -1e30 handle the pad mask).
// ---------------------------------------------------------------------------
__global__ __launch_bounds__(32)
void k_attn(const unsigned short* __restrict__ qbuf,
            const unsigned short* __restrict__ kbuf,
            const unsigned short* __restrict__ vtbuf,   // [b,h,hd,key]
            const unsigned char* __restrict__ binsT,    // [b,key,row]
            const float* __restrict__ dbias,
            unsigned short* __restrict__ obuf) {
    __shared__ unsigned short Sb[16 * NNTOK];   // bf16 score rows (32 KB)
    __shared__ float db[32];
    __shared__ float red[32];
    const int lane = threadIdx.x;
    const int half = lane >> 4;
    const int lidx = lane & 15;
    const int bid = blockIdx.x;             // B*H*(N/16) = 4096
    const int q16 = bid & 63;
    const int h   = (bid >> 6) & 7;
    const int b   = bid >> 9;
    db[lane] = (lane < NBINS) ? dbias[h * NBINS + lane] : -1e30f;
    const unsigned short* Q  = qbuf + (((size_t)(b * HH + h)) * NNTOK + q16 * 16) * HDIM;
    const unsigned short* Km = kbuf + ((size_t)(b * HH + h)) * NNTOK * HDIM;
    const unsigned short* Vt = vtbuf + ((size_t)(b * HH + h)) * HDIM * NNTOK;
    const unsigned char*  Bt = binsT + (size_t)b * NNTOK * NNTOK + q16 * 16 + 8 * half;

    // Q fragments (A 16x64 = two K=32 fragments), resident in VGPRs
    v16bf qf[2];
#pragma unroll
    for (int f = 0; f < 2; ++f) {
        const unsigned short* qp = Q + lidx * HDIM + 32 * f + 16 * half;
#pragma unroll
        for (int j = 0; j < 8; ++j) qf[f][j] = bfbits(qp[j]);
#pragma unroll
        for (int j = 0; j < 8; ++j) qf[f][8 + j] = bfbits(qp[8 + j]);
    }
    __syncthreads();

    const float scale = 0.125f;  // 1/sqrt(64)
    for (int kt = 0; kt < 64; ++kt) {
        const int key = kt * 16 + lidx;
        v16bf kf0, kf1;   // B = K^T: lane holds key column, elems walk hd
        const unsigned short* kp0 = Km + (size_t)key * HDIM + 16 * half;
        const unsigned short* kp1 = kp0 + 32;
#pragma unroll
        for (int j = 0; j < 16; ++j) { kf0[j] = bfbits(kp0[j]); kf1[j] = bfbits(kp1[j]); }
        v8f s;
#pragma unroll
        for (int r = 0; r < 8; ++r) s[r] = 0.0f;
        s = __builtin_amdgcn_wmma_f32_16x16x32_bf16(false, qf[0], false, kf0, (short)0, s, false, false);
        s = __builtin_amdgcn_wmma_f32_16x16x32_bf16(false, qf[1], false, kf1, (short)0, s, false, false);
        // 8 bins for this lane's 8 rows: one aligned 8-byte load
        const unsigned long long bw = *(const unsigned long long*)(Bt + (size_t)key * NNTOK);
#pragma unroll
        for (int r = 0; r < 8; ++r) {
            const float bv = db[(bw >> (8 * r)) & 0xffu];
            Sb[(r + 8 * half) * NNTOK + key] = f2bf(s[r] * scale + bv);
        }
    }
    __syncthreads();

    // softmax: 2 lanes per row (row = lane&15, segment = lane>>4), 512 cols each
    unsigned short* Srow = &Sb[lidx * NNTOK + half * 512];
    float mx = -1e30f;
    for (int c = 0; c < 512; ++c) mx = fmaxf(mx, bf2f(Srow[c]));
    red[lane] = mx;
    __syncthreads();
    const float rmx = fmaxf(red[lidx], red[lidx + 16]);
    __syncthreads();
    float sum = 0.0f;
    for (int c = 0; c < 512; ++c) {
        const unsigned short eb = f2bf(__expf(bf2f(Srow[c]) - rmx));
        Srow[c] = eb;
        sum += bf2f(eb);
    }
    red[lane] = sum;
    __syncthreads();
    const float inv = 1.0f / (red[lidx] + red[lidx + 16]);
    for (int c = 0; c < 512; ++c) Srow[c] = f2bf(bf2f(Srow[c]) * inv);
    __syncthreads();

    // O = P @ V : 1024 keys in 32 chunks of 32; V fragments contiguous via Vt
    v8f o[4];
#pragma unroll
    for (int ni = 0; ni < 4; ++ni)
#pragma unroll
        for (int r = 0; r < 8; ++r) o[ni][r] = 0.0f;

    for (int kc = 0; kc < 32; ++kc) {
        const int kb0 = kc * 32;
        v16bf pf;   // A fragment straight out of LDS score buffer
        const unsigned short* pp = &Sb[lidx * NNTOK + kb0 + 16 * half];
#pragma unroll
        for (int j = 0; j < 8; ++j) pf[j] = bfbits(pp[j]);
#pragma unroll
        for (int j = 0; j < 8; ++j) pf[8 + j] = bfbits(pp[8 + j]);
#pragma unroll
        for (int ni = 0; ni < 4; ++ni) {
            const unsigned short* vp = Vt + (size_t)(ni * 16 + lidx) * NNTOK + kb0 + 16 * half;
            v16bf vf;  // B[k=key][n=hd]: lane holds hd col, elems walk keys
#pragma unroll
            for (int j = 0; j < 16; ++j) vf[j] = bfbits(vp[j]);
            o[ni] = __builtin_amdgcn_wmma_f32_16x16x32_bf16(
                false, pf, false, vf, (short)0, o[ni], false, false);
        }
    }

#pragma unroll
    for (int ni = 0; ni < 4; ++ni)
#pragma unroll
        for (int r = 0; r < 8; ++r) {
            const int qrow = q16 * 16 + r + 8 * half;
            obuf[((size_t)(b * NNTOK + qrow)) * DD + h * HDIM + ni * 16 + lidx] = f2bf(o[ni][r]);
        }
}

// ---------------------------------------------------------------------------
// launch
// ---------------------------------------------------------------------------
extern "C" void kernel_launch(void* const* d_in, const int* in_sizes, int n_in,
                              void* d_out, int out_size, void* d_ws, size_t ws_size,
                              hipStream_t stream) {
    (void)in_sizes; (void)n_in; (void)out_size; (void)ws_size;
    const float* slot_features = (const float*)d_in[0];
    const float* dist          = (const float*)d_in[1];
    const unsigned char* mask  = (const unsigned char*)d_in[2];
    const float* in_w   = (const float*)d_in[3];
    const float* in_b   = (const float*)d_in[4];
    const float* in_g   = (const float*)d_in[5];
    const float* in_be  = (const float*)d_in[6];
    const float* dbias  = (const float*)d_in[7];
    const float* qkv_w  = (const float*)d_in[8];
    const float* qkv_b  = (const float*)d_in[9];
    const float* out_w  = (const float*)d_in[10];
    const float* out_b  = (const float*)d_in[11];
    const float* ffn_w1 = (const float*)d_in[12];
    const float* ffn_b1 = (const float*)d_in[13];
    const float* ffn_w2 = (const float*)d_in[14];
    const float* ffn_b2 = (const float*)d_in[15];
    const float* n1_g   = (const float*)d_in[16];
    const float* n1_b   = (const float*)d_in[17];
    const float* n2_g   = (const float*)d_in[18];
    const float* n2_b   = (const float*)d_in[19];
    const float* op_w   = (const float*)d_in[20];
    const float* op_b   = (const float*)d_in[21];

    char* ws = (char*)d_ws;
    size_t off = 0;
    auto alloc = [&](size_t bytes) -> char* {
        char* p = ws + off;
        off += (bytes + 255) & ~(size_t)255;
        return p;
    };
    float*          h    = (float*)alloc((size_t)MROWS * DD * 4);
    unsigned short* hnb  = (unsigned short*)alloc((size_t)MROWS * DD * 2);
    unsigned short* qbf  = (unsigned short*)alloc((size_t)MROWS * DD * 2);
    unsigned short* kbf  = (unsigned short*)alloc((size_t)MROWS * DD * 2);
    unsigned short* vtbf = (unsigned short*)alloc((size_t)MROWS * DD * 2);
    unsigned short* obf  = (unsigned short*)alloc((size_t)MROWS * DD * 2);
    unsigned short* a1   = (unsigned short*)alloc((size_t)MROWS * DFFN * 2);
    unsigned char*  bins = (unsigned char*)alloc((size_t)BBAT * NNTOK * NNTOK);
    unsigned int*   bmax = (unsigned int*)alloc(256);
    unsigned short* wqkv = (unsigned short*)alloc((size_t)LLAY * DD * 3 * DD * 2);
    unsigned short* wout = (unsigned short*)alloc((size_t)LLAY * DD * DD * 2);
    unsigned short* wf1  = (unsigned short*)alloc((size_t)LLAY * DD * DFFN * 2);
    unsigned short* wf2  = (unsigned short*)alloc((size_t)LLAY * DFFN * DD * 2);
    unsigned short* wop  = (unsigned short*)alloc((size_t)DD * DD * 2);

    // weights -> bf16, transposed to [Ncol][K] per layer
    k_cvtT<<<dim3(DD / 32, (3 * DD) / 32, LLAY), 256, 0, stream>>>(qkv_w, wqkv, DD, 3 * DD);
    k_cvtT<<<dim3(DD / 32, DD / 32, LLAY), 256, 0, stream>>>(out_w, wout, DD, DD);
    k_cvtT<<<dim3(DD / 32, DFFN / 32, LLAY), 256, 0, stream>>>(ffn_w1, wf1, DD, DFFN);
    k_cvtT<<<dim3(DFFN / 32, DD / 32, LLAY), 256, 0, stream>>>(ffn_w2, wf2, DFFN, DD);
    k_cvtT<<<dim3(DD / 32, DD / 32, 1), 256, 0, stream>>>(op_w, wop, DD, DD);

    k_inproj<<<MROWS, 256, 0, stream>>>(slot_features, in_w, in_b, in_g, in_be, h);

    k_maxinit<<<1, 32, 0, stream>>>(bmax);
    k_max<<<dim3(64, BBAT), 256, 0, stream>>>(dist, bmax);
    k_binsT<<<dim3(NNTOK / 64, NNTOK / 64, BBAT), 256, 0, stream>>>(dist, bmax, mask, bins);

    for (int i = 0; i < LLAY; ++i) {
        k_ln<<<MROWS, 256, 0, stream>>>(h, n1_g + i * DD, n1_b + i * DD, hnb);
        k_gemm<EPI_QKV><<<dim3((3 * DD) / BN, MROWS / BM), 256, 0, stream>>>(
            hnb, wqkv + (size_t)i * DD * 3 * DD, qkv_b + i * 3 * DD, 3 * DD, DD,
            nullptr, nullptr, qbf, kbf, vtbf);
        k_attn<<<BBAT * HH * (NNTOK / 16), 32, 0, stream>>>(
            qbf, kbf, vtbf, bins, dbias, obf);
        k_gemm<EPI_RES><<<dim3(DD / BN, MROWS / BM), 256, 0, stream>>>(
            obf, wout + (size_t)i * DD * DD, out_b + i * DD, DD, DD,
            h, nullptr, nullptr, nullptr, nullptr);
        k_ln<<<MROWS, 256, 0, stream>>>(h, n2_g + i * DD, n2_b + i * DD, hnb);
        k_gemm<EPI_GELU><<<dim3(DFFN / BN, MROWS / BM), 256, 0, stream>>>(
            hnb, wf1 + (size_t)i * DD * DFFN, ffn_b1 + i * DFFN, DFFN, DD,
            nullptr, a1, nullptr, nullptr, nullptr);
        k_gemm<EPI_RES><<<dim3(DD / BN, MROWS / BM), 256, 0, stream>>>(
            a1, wf2 + (size_t)i * DFFN * DD, ffn_b2 + i * DD, DD, DFFN,
            h, nullptr, nullptr, nullptr, nullptr);
    }

    k_cvt<<<(MROWS * DD + 255) / 256, 256, 0, stream>>>(h, hnb, MROWS * DD);
    k_gemm<EPI_OUT><<<dim3(DD / BN, MROWS / BM), 256, 0, stream>>>(
        hnb, wop, op_b, DD, DD, (float*)d_out, nullptr, nullptr, nullptr, nullptr);
}
